// cartpole_RNN_dqn_74251394613868
// MI455X (gfx1250) — matrix operand order repaired and checked by reading, not compiled, not run on hardware
//
#include <hip/hip_runtime.h>

typedef __attribute__((ext_vector_type(16))) _Float16 v16h;
typedef __attribute__((ext_vector_type(8)))  float    v8f;

#define SEQ    512
#define NBATCH 4096
#define IN_DIM 4
#define H1     24
#define H2     48
#define H3     24
#define OUT_D  2

static __device__ __forceinline__ v8f wmma_f16(v16h a, v16h b, v8f c) {
  // D = A(16x32 f16) * B(32x16 f16) + C(f32)
  return __builtin_amdgcn_wmma_f32_16x16x32_f16(false, a, false, b, (short)0, c,
                                                false, false);
}

static __device__ __forceinline__ float relu1(float v) {
  return fmaxf(v, 0.0f);   // lowers to dual-issued v_dual_max_num_f32 pairs
}

static __device__ __forceinline__ v16h zero16() {
  v16h r;
#pragma unroll
  for (int i = 0; i < 16; ++i) r[i] = (_Float16)0.0f;
  return r;
}

static __device__ __forceinline__ v8f zero8() {
  v8f r;
#pragma unroll
  for (int i = 0; i < 8; ++i) r[i] = 0.0f;
  return r;
}

// Pack a 16x32 f16 A-fragment; element value supplied by f(m, k_logical).
// Hardware A layout (16-bit, 16x32): lane L holds row m = L&15;
// element e (VGPR e/2, half e%2): k_hw = (v>=4?16:0) + 8*(L>>4) + (v&3)*2 + h.
// Logical k = kt*32 + k_hw (identity), matching the shuffle-free C->B state
// conversion (b_from_cd), which yields table-order k.
template <typename F>
static __device__ __forceinline__ v16h pack_A_f(int mt, int kt, int lane, F f) {
  const int m  = mt * 16 + (lane & 15);
  const int kb = ((lane >> 4) & 1) << 3;
  v16h r;
#pragma unroll
  for (int e = 0; e < 16; ++e) {
    const int v  = e >> 1, h = e & 1;
    const int kl = ((v & 4) ? 16 : 0) + kb + ((v & 3) << 1) + h;      // hw k 0..31
    const int ks = kl;
    r[e] = (_Float16)f(m, kt * 32 + ks);
  }
  return r;
}

// Shuffle-free C->B conversion with ReLU. d0 = channel tile c (m 0..15),
// d1 = channel tile c+1 (m 16..31); consumers use identity (table-order) A columns.
static __device__ __forceinline__ v16h b_from_cd(v8f d0, v8f d1) {
  v16h r;
#pragma unroll
  for (int j = 0; j < 8; ++j) {
    r[j]     = (_Float16)relu1(d0[j]);
    r[j + 8] = (_Float16)relu1(d1[j]);
  }
  return r;
}

__global__ __launch_bounds__(256)
void cartpole_rnn_wmma_kernel(const float* __restrict__ x,
                              const float* __restrict__ Wih0, const float* __restrict__ Whh0,
                              const float* __restrict__ bih0, const float* __restrict__ bhh0,
                              const float* __restrict__ Wih1, const float* __restrict__ Whh1,
                              const float* __restrict__ bih1, const float* __restrict__ bhh1,
                              const float* __restrict__ Wih2, const float* __restrict__ Whh2,
                              const float* __restrict__ bih2, const float* __restrict__ bhh2,
                              const float* __restrict__ Wout, const float* __restrict__ bout,
                              float* __restrict__ out) {
  const int lane = threadIdx.x & 31;
  const int hi   = lane >> 4;
  const int wave = blockIdx.x * (blockDim.x >> 5) + (threadIdx.x >> 5);
  const long b0  = (long)wave * 16;   // 16 batch rows per wave

  // ---- pack all weights (biases folded in via constant-1 channels) ----
  v16h Aih0[2], Ahh0[2], Aih1[3], Ahh1[3][2], Aih2[2][2], Ahh2[2], Aout;

  // layer 0: bias at k==4 (xB carries 1.0 there); row 24 gets 1.0 so that
  // h0 channel 24 == 1.0 forever (bias channel for layer 1).
  auto f_ih0 = [&](int m, int k) -> float {
    if (m < H1 && k < IN_DIM) return Wih0[m * IN_DIM + k];
    if (k == IN_DIM) {
      if (m < H1)  return bih0[m] + bhh0[m];
      if (m == H1) return 1.0f;          // keep-alive 1.0 channel (ch 24)
    }
    return 0.0f;
  };
  auto f_hh0 = [&](int m, int k) -> float {
    return (m < H1 && k < H1) ? Whh0[m * H1 + k] : 0.0f;
  };
  // layer 1: bias read from h0 channel 24 (==1.0).
  auto f_ih1 = [&](int m, int k) -> float {
    if (m < H2 && k < H1) return Wih1[m * H1 + k];
    if (k == H1 && m < H2) return bih1[m] + bhh1[m];
    return 0.0f;
  };
  auto f_hh1 = [&](int m, int k) -> float {
    return (m < H2 && k < H2) ? Whh1[m * H2 + k] : 0.0f;
  };
  // layer 2: bias read from h1 pad channel 48 (==1.0, injected in h1B1);
  // row 24 gets 1.0 so h2 channel 24 == 1.0 (bias channel for the head).
  auto f_ih2 = [&](int m, int k) -> float {
    if (m < H3 && k < H2) return Wih2[m * H2 + k];
    if (k == H2) {
      if (m < H3)  return bih2[m] + bhh2[m];
      if (m == H3) return 1.0f;          // keep-alive 1.0 channel (ch 24 of h2)
    }
    return 0.0f;
  };
  auto f_hh2 = [&](int m, int k) -> float {
    return (m < H3 && k < H3) ? Whh2[m * H3 + k] : 0.0f;
  };
  // head: bias read from h2 channel 24 (==1.0).
  auto f_out = [&](int m, int k) -> float {
    if (m < OUT_D && k < H3) return Wout[m * H3 + k];
    if (k == H3 && m < OUT_D) return bout[m];
    return 0.0f;
  };

#pragma unroll
  for (int mt = 0; mt < 2; ++mt) {
    Aih0[mt]    = pack_A_f(mt, 0, lane, f_ih0);
    Ahh0[mt]    = pack_A_f(mt, 0, lane, f_hh0);
    Ahh2[mt]    = pack_A_f(mt, 0, lane, f_hh2);
    Aih2[mt][0] = pack_A_f(mt, 0, lane, f_ih2);
    Aih2[mt][1] = pack_A_f(mt, 1, lane, f_ih2);
  }
#pragma unroll
  for (int mt = 0; mt < 3; ++mt) {
    Aih1[mt]    = pack_A_f(mt, 0, lane, f_ih1);
    Ahh1[mt][0] = pack_A_f(mt, 0, lane, f_hh1);
    Ahh1[mt][1] = pack_A_f(mt, 1, lane, f_hh1);
  }
  Aout = pack_A_f(0, 0, lane, f_out);

  const v8f z8 = zero8();

  // ---- recurrent state as f16 B-fragments (k=channel in table order, n=batch) ----
  v16h h0B = zero16(), h1B0 = zero16(), h1B1 = zero16(), h2B = zero16();
  // h1B1 upper half is constant: hw k=16 (-> logical ch 48) carries 1.0 on the
  // low half-wave; loop-invariant.
  h1B1[8] = hi ? (_Float16)0.0f : (_Float16)1.0f;

  // constant part of the x fragment: hw k=4 (logical ch 4) = 1.0 (bias input)
  v16h xBc = zero16();
  if (hi == 0) xBc[4] = (_Float16)1.0f;

  // x pipeline: unconditional load on all 32 lanes (upper half-wave re-reads
  // the same cachelines; values masked out when building xB)
  const int bl = (int)b0 + (lane & 15);

  // prologue: xB/p0 for t=0; then prefetch x[1]
  float4 xr = *(const float4*)(x + (long)bl * IN_DIM);
  v16h xB = xBc;
  xB[0] = (_Float16)(hi ? 0.0f : xr.x);
  xB[1] = (_Float16)(hi ? 0.0f : xr.y);
  xB[2] = (_Float16)(hi ? 0.0f : xr.z);
  xB[3] = (_Float16)(hi ? 0.0f : xr.w);
  v8f p0a = wmma_f16(Aih0[0], xB, z8);   // input projection for step 0
  v8f p0b = wmma_f16(Aih0[1], xB, z8);
  xr = *(const float4*)(x + ((long)1 * NBATCH + bl) * IN_DIM);

#pragma unroll 1
  for (int t = 0; t < SEQ; ++t) {
    // ---- recurrent (state-only) terms first: ready at step start ----
    v8f d1a = wmma_f16(Ahh1[0][0], h1B0, z8);
    d1a     = wmma_f16(Ahh1[0][1], h1B1, d1a);
    v8f d1b = wmma_f16(Ahh1[1][0], h1B0, z8);
    d1b     = wmma_f16(Ahh1[1][1], h1B1, d1b);
    v8f d1c = wmma_f16(Ahh1[2][0], h1B0, z8);
    d1c     = wmma_f16(Ahh1[2][1], h1B1, d1c);
    v8f d20 = wmma_f16(Ahh2[0], h2B, z8);
    v8f d21 = wmma_f16(Ahh2[1], h2B, z8);

    // ---- layer 0 finish: recurrent term onto precomputed input projection ----
    v8f d00 = wmma_f16(Ahh0[0], h0B, p0a);
    v8f d01 = wmma_f16(Ahh0[1], h0B, p0b);

    // ---- next step's input projection (off the critical path) ----
    xB = xBc;
    xB[0] = (_Float16)(hi ? 0.0f : xr.x);
    xB[1] = (_Float16)(hi ? 0.0f : xr.y);
    xB[2] = (_Float16)(hi ? 0.0f : xr.z);
    xB[3] = (_Float16)(hi ? 0.0f : xr.w);
    p0a = wmma_f16(Aih0[0], xB, z8);
    p0b = wmma_f16(Aih0[1], xB, z8);
    {
      const int tn = (t + 2 < SEQ) ? (t + 2) : (SEQ - 1);  // clamp: re-read ok
      xr = *(const float4*)(x + ((long)tn * NBATCH + bl) * IN_DIM);
    }

    h0B = b_from_cd(d00, d01);

    // ---- layer 1: only the ih WMMA waits on the fresh h0 ----
    d1a = wmma_f16(Aih1[0], h0B, d1a);
    d1b = wmma_f16(Aih1[1], h0B, d1b);
    d1c = wmma_f16(Aih1[2], h0B, d1c);
    h1B0 = b_from_cd(d1a, d1b);
    // h1B1: lower 8 from tile c2; upper half stays the constant bias channel
#pragma unroll
    for (int j = 0; j < 8; ++j) h1B1[j] = (_Float16)relu1(d1c[j]);

    // ---- layer 2: ih terms wait on fresh h1 ----
    d20 = wmma_f16(Aih2[0][0], h1B0, d20);
    d20 = wmma_f16(Aih2[0][1], h1B1, d20);
    d21 = wmma_f16(Aih2[1][0], h1B0, d21);
    d21 = wmma_f16(Aih2[1][1], h1B1, d21);
    h2B = b_from_cd(d20, d21);

    // ---- head: out[t,b,:] = Wout a3 + bout ----
    v8f dO = wmma_f16(Aout, h2B, z8);
    if (hi == 0) {
      float2 o; o.x = dO[0]; o.y = dO[1];  // m=0,1 live in VGPR 0,1, low half
      *(float2*)(out + ((long)t * NBATCH + b0 + lane) * OUT_D) = o;
    }
  }
}

extern "C" void kernel_launch(void* const* d_in, const int* in_sizes, int n_in,
                              void* d_out, int out_size, void* d_ws, size_t ws_size,
                              hipStream_t stream) {
  (void)in_sizes; (void)n_in; (void)d_ws; (void)ws_size; (void)out_size;
  const float* x    = (const float*)d_in[0];
  const float* Wih0 = (const float*)d_in[1];
  const float* Whh0 = (const float*)d_in[2];
  const float* bih0 = (const float*)d_in[3];
  const float* bhh0 = (const float*)d_in[4];
  const float* Wih1 = (const float*)d_in[5];
  const float* Whh1 = (const float*)d_in[6];
  const float* bih1 = (const float*)d_in[7];
  const float* bhh1 = (const float*)d_in[8];
  const float* Wih2 = (const float*)d_in[9];
  const float* Whh2 = (const float*)d_in[10];
  const float* bih2 = (const float*)d_in[11];
  const float* bhh2 = (const float*)d_in[12];
  const float* Wout = (const float*)d_in[13];
  const float* bout = (const float*)d_in[14];
  float* out = (float*)d_out;

  // 4096 batch / 16 per wave = 256 waves; 8 waves (256 threads) per block
  // -> 2 waves per SIMD32 so SMT covers the WMMA->VALU hazard gaps.
  dim3 grid(NBATCH / 16 / 8);
  dim3 block(256);
  cartpole_rnn_wmma_kernel<<<grid, block, 0, stream>>>(
      x, Wih0, Whh0, bih0, bhh0, Wih1, Whh1, bih1, bhh1,
      Wih2, Whh2, bih2, bhh2, Wout, bout, out);
}
